// ConvLSTMSeq2Seq_48962627175231
// MI455X (gfx1250) — matrix operand order repaired and checked by reading, not compiled, not run on hardware
//
#include <hip/hip_runtime.h>
#include <hip/hip_bf16.h>

typedef __attribute__((ext_vector_type(16))) _Float16 v16h;
typedef __attribute__((ext_vector_type(8)))  float    v8f;

#define TT 10   // time steps
#define NB 8    // batch
#define HID 64
#define HW 64

// ---------------------------------------------------------------------------
// Swizzle f32 conv weights (4*HID, inK, 3, 3) into WMMA A-fragment layout:
//   frag[t9][kk][mt][lane][hi]  (f16), K padded to ksteps*32 with zeros.
// Per-lane k order used here for 16x32 f16 A (contiguous, B reads match):
//   lanes 0-15 : M=lane,    K = hi         (0..15)
//   lanes 16-31: M=lane-16, K = hi+16      (16..31)
// ---------------------------------------------------------------------------
__global__ __launch_bounds__(256) void convert_wfrag(
    const float* __restrict__ W, _Float16* __restrict__ dst, int inK, int ksteps)
{
    int idx = blockIdx.x * 256 + threadIdx.x;
    int total = 9 * ksteps * 8192;
    if (idx >= total) return;
    int hi   = idx & 15;
    int lane = (idx >> 4) & 31;
    int mt   = (idx >> 9) & 15;
    int rem  = idx >> 13;
    int kk   = rem % ksteps;
    int t9   = rem / ksteps;
    int m    = mt * 16 + (lane & 15);
    int k32  = (lane < 16) ? hi : hi + 16;
    int k    = kk * 32 + k32;
    int ky   = t9 / 3, kx = t9 - ky * 3;
    float v  = 0.f;
    if (k < inK) v = W[((m * inK + k) * 3 + ky) * 3 + kx];
    dst[idx] = (_Float16)v;
}

// ---------------------------------------------------------------------------
// One ConvLSTM cell step for one (batch b, output row y) per block.
// Implicit GEMM: M=256 gate channels, N=64 pixels of the row, K=inK*9.
// 8 waves x (2 M-tiles x 4 N-tiles) 16x16 accumulators.
// ---------------------------------------------------------------------------
__global__ __launch_bounds__(256) void cell_kernel(
    const float*    __restrict__ xsrc_f32, int t, int inK_x, int ks_x,
    const _Float16* __restrict__ xsrc_h,     // (B,H,W,64) f16 or null
    const _Float16* __restrict__ hsrc,       // recurrent h (B,H,W,64) f16
    const _Float16* __restrict__ AfragX,
    const _Float16* __restrict__ AfragH,
    const float*    __restrict__ Wc,         // (4,64,64,64)
    const float*    __restrict__ bias,       // (4,64,64,64)
    float*          __restrict__ c,          // (B,64,64,64) in-place
    _Float16*       __restrict__ hout)       // (B,H,W,64) f16
{
    extern __shared__ char smem[];           // 64 KB: staging then exchange
    _Float16* lds_h = (_Float16*)smem;
    float*    pre   = (float*)smem;

    const int tid  = threadIdx.x;
    const int lane = tid & 31;
    const int wave = tid >> 5;
    const int b    = blockIdx.x >> 6;
    const int y    = blockIdx.x & 63;

    v8f acc[2][4];
#pragma unroll
    for (int i = 0; i < 2; ++i)
#pragma unroll
        for (int j = 0; j < 4; ++j) acc[i][j] = (v8f){};

    for (int src = 0; src < 2; ++src) {
        const int ks   = src ? 2 : ks_x;
        const int inKp = ks * 32;
        const int inK  = src ? HID : inK_x;
        const _Float16* Af = src ? AfragH : AfragX;
        const _Float16* hs = src ? hsrc : xsrc_h;

        // --- stage 3 halo rows into LDS as f16 [dy][px(66)][ch(inKp)] ---
        const int total = 3 * 66 * inKp;
        for (int e = tid; e < total; e += 256) {
            int ch  = e % inKp;
            int px  = (e / inKp) % 66;
            int dy  = e / (inKp * 66);
            int row = y + dy - 1, gx = px - 1;
            float v = 0.f;
            if ((unsigned)row < 64u && (unsigned)gx < 64u && ch < inK) {
                if (src == 0 && xsrc_f32)   // encoder-0 input x: (B,T,1,H,W)
                    v = xsrc_f32[(b * TT + t) * 4096 + row * 64 + gx];
                else
                    v = (float)hs[((b * 64 + row) * 64 + gx) * 64 + ch];
            }
            lds_h[e] = (_Float16)v;
        }
        __syncthreads();

        // --- implicit GEMM over 9 taps x ks K-steps ---
        const int nl = lane & 15;
        const int kh = (lane >> 4) * 16;
        for (int t9 = 0; t9 < 9; ++t9) {
            const int ky = t9 / 3, kx = t9 - ky * 3;
            for (int kk = 0; kk < ks; ++kk) {
                const v16h a0 = *(const v16h*)(Af +
                    ((((t9 * ks + kk) * 16 + wave * 2)     * 32 + lane) * 16));
                const v16h a1 = *(const v16h*)(Af +
                    ((((t9 * ks + kk) * 16 + wave * 2 + 1) * 32 + lane) * 16));
#pragma unroll
                for (int nt = 0; nt < 4; ++nt) {
                    const int px = nt * 16 + nl + kx;
                    const v16h bf = *(const v16h*)(lds_h +
                        (ky * 66 + px) * inKp + kk * 32 + kh);
                    acc[0][nt] = __builtin_amdgcn_wmma_f32_16x16x32_f16(
                        false, a0, false, bf, (short)0, acc[0][nt], false, false);
                    acc[1][nt] = __builtin_amdgcn_wmma_f32_16x16x32_f16(
                        false, a1, false, bf, (short)0, acc[1][nt], false, false);
                }
            }
        }
        __syncthreads();
    }

    // --- exchange accumulators through LDS: pre[m(256)][n(64)] f32 ---
    {
        const int rowsel = (lane >> 4) * 8;
        const int ncol   = lane & 15;
#pragma unroll
        for (int mi = 0; mi < 2; ++mi) {
            const int mbase = (wave * 2 + mi) * 16 + rowsel;
#pragma unroll
            for (int nt = 0; nt < 4; ++nt) {
                const int nbase = nt * 16 + ncol;
#pragma unroll
                for (int j = 0; j < 8; ++j)
                    pre[(mbase + j) * 64 + nbase] = acc[mi][nt][j];
            }
        }
    }
    __syncthreads();

    // --- gate math + state update (f32), h out as f16 (B,H,W,C) ---
    for (int it = 0; it < 16; ++it) {
        int p  = tid + it * 256;        // 0..4095 = (ch, n)
        int ch = p >> 6, n = p & 63;
        int sp = y * 64 + n;
        float cold = c[(b * 64 + ch) * 4096 + sp];
        float pg[4];
#pragma unroll
        for (int g = 0; g < 4; ++g)
            pg[g] = pre[(g * 64 + ch) * 64 + n]
                  + Wc[(g * 64 + ch) * 4096 + sp] * cold
                  + bias[(g * 64 + ch) * 4096 + sp];
        float i_ = 1.f / (1.f + __expf(-pg[0]));
        float f_ = 1.f / (1.f + __expf(-pg[1]));
        float g_ = tanhf(pg[2]);
        float o_ = 1.f / (1.f + __expf(-pg[3]));
        float cn = f_ * cold + i_ * g_;
        float hn = o_ * tanhf(cn);
        c[(b * 64 + ch) * 4096 + sp] = cn;
        hout[((b * 64 + y) * 64 + n) * 64 + ch] = (_Float16)hn;
    }
}

// ---------------------------------------------------------------------------
// Final 1-channel 3x3 conv + sigmoid per decoded frame (negligible FLOPs).
// ---------------------------------------------------------------------------
__global__ __launch_bounds__(256) void final_conv(
    const _Float16* __restrict__ h,       // (B,H,W,64) f16
    const float* __restrict__ fw,         // (1,64,3,3)
    const float* __restrict__ fb,
    float* __restrict__ out, int frame)
{
    int gid = blockIdx.x * 256 + threadIdx.x;   // 0..32767
    int b = gid >> 12, sp = gid & 4095;
    int y = sp >> 6, x = sp & 63;
    float s = fb[0];
    for (int ky = 0; ky < 3; ++ky) {
        int row = y + ky - 1;
        if ((unsigned)row >= 64u) continue;
        for (int kx = 0; kx < 3; ++kx) {
            int col = x + kx - 1;
            if ((unsigned)col >= 64u) continue;
            const _Float16* hp = h + ((b * 64 + row) * 64 + col) * 64;
#pragma unroll 8
            for (int ch = 0; ch < 64; ++ch)
                s += (float)hp[ch] * fw[ch * 9 + ky * 3 + kx];
        }
    }
    out[(b * TT + frame) * 4096 + sp] = 1.f / (1.f + __expf(-s));
}

// ---------------------------------------------------------------------------
extern "C" void kernel_launch(void* const* d_in, const int* in_sizes, int n_in,
                              void* d_out, int out_size, void* d_ws, size_t ws_size,
                              hipStream_t stream)
{
    (void)in_sizes; (void)n_in; (void)out_size; (void)ws_size;
    const float* x = (const float*)d_in[0];
    // per cell: Wx, Wh, Wc, b   (cells: enc0, enc1, dec0, dec1)
    const float* W[4][4];
    for (int cidx = 0; cidx < 4; ++cidx)
        for (int j = 0; j < 4; ++j)
            W[cidx][j] = (const float*)d_in[1 + cidx * 4 + j];
    const float* fin_w = (const float*)d_in[17];
    const float* fin_b = (const float*)d_in[18];
    // prediction_len is fixed at 10 by the reference setup.

    char* ws = (char*)d_ws;
    size_t off = 0;
    auto take = [&](size_t bytes) -> char* {
        char* p = ws + off;
        off = (off + bytes + 255) & ~(size_t)255;
        return p;
    };

    const int inKx[4] = {1, 64, 64, 64};
    const int ksx[4]  = {1, 2, 2, 2};
    _Float16* fragX[4]; _Float16* fragH[4];
    for (int cidx = 0; cidx < 4; ++cidx) {
        fragX[cidx] = (_Float16*)take((size_t)9 * ksx[cidx] * 8192 * sizeof(_Float16));
        fragH[cidx] = (_Float16*)take((size_t)9 * 2 * 8192 * sizeof(_Float16));
    }
    const size_t hbytes = (size_t)NB * 64 * 64 * 64 * sizeof(_Float16);
    const size_t cbytes = (size_t)NB * 64 * 64 * 64 * sizeof(float);
    _Float16* hping[4]; _Float16* hpong[4]; float* cbuf[4];
    for (int cidx = 0; cidx < 4; ++cidx) {
        hping[cidx] = (_Float16*)take(hbytes);
        hpong[cidx] = (_Float16*)take(hbytes);
        cbuf[cidx]  = (float*)take(cbytes);
    }

    // Swizzle weights into WMMA fragment layout (amortized per launch).
    for (int cidx = 0; cidx < 4; ++cidx) {
        int nx = 9 * ksx[cidx] * 8192;
        convert_wfrag<<<(nx + 255) / 256, 256, 0, stream>>>(
            W[cidx][0], fragX[cidx], inKx[cidx], ksx[cidx]);
        int nh = 9 * 2 * 8192;
        convert_wfrag<<<(nh + 255) / 256, 256, 0, stream>>>(
            W[cidx][1], fragH[cidx], 64, 2);
    }
    // Zero initial states.
    for (int cidx = 0; cidx < 4; ++cidx) {
        hipMemsetAsync(hping[cidx], 0, hbytes, stream);
        hipMemsetAsync(cbuf[cidx], 0, cbytes, stream);
    }

    const size_t shmem = 65536;

    // Encoder: 10 time steps, 2 stacked cells.
    for (int t = 0; t < TT; ++t) {
        cell_kernel<<<NB * 64, 256, shmem, stream>>>(
            x, t, 1, 1, (const _Float16*)nullptr, hping[0],
            fragX[0], fragH[0], W[0][2], W[0][3], cbuf[0], hpong[0]);
        cell_kernel<<<NB * 64, 256, shmem, stream>>>(
            (const float*)nullptr, 0, 64, 2, hpong[0], hping[1],
            fragX[1], fragH[1], W[1][2], W[1][3], cbuf[1], hpong[1]);
        { _Float16* tmp = hping[0]; hping[0] = hpong[0]; hpong[0] = tmp; }
        { _Float16* tmp = hping[1]; hping[1] = hpong[1]; hpong[1] = tmp; }
    }

    // Decoder: feed top encoder hidden, autoregress for 10 frames.
    const _Float16* s = hping[1];
    for (int f = 0; f < TT; ++f) {
        cell_kernel<<<NB * 64, 256, shmem, stream>>>(
            (const float*)nullptr, 0, 64, 2, s, hping[2],
            fragX[2], fragH[2], W[2][2], W[2][3], cbuf[2], hpong[2]);
        cell_kernel<<<NB * 64, 256, shmem, stream>>>(
            (const float*)nullptr, 0, 64, 2, hpong[2], hping[3],
            fragX[3], fragH[3], W[3][2], W[3][3], cbuf[3], hpong[3]);
        final_conv<<<(NB * 4096) / 256, 256, 0, stream>>>(
            hpong[3], fin_w, fin_b, (float*)d_out, f);
        s = hpong[3];
        { _Float16* tmp = hping[2]; hping[2] = hpong[2]; hpong[2] = tmp; }
        { _Float16* tmp = hping[3]; hping[3] = hpong[3]; hpong[3] = tmp; }
    }
}
